// FFTBlock_1d_conv1x1width_resize_52080773432007
// MI455X (gfx1250) — matrix operand order repaired and checked by reading, not compiled, not run on hardware
//
#include <hip/hip_runtime.h>
#include <hip/hip_bf16.h>

// ---------------------------------------------------------------------------
// FFTBlock: spectral branch folded into two f16 WMMA GEMMs via precomputed
// M1 = W_spec @ rfft-basis and M2 = irfft-basis; spatial (7,1)-strided conv
// as a third WMMA GEMM. fp32 accumulation throughout (v_wmma_f32_16x16x32_f16).
//
// Round-3 fix: __builtin_amdgcn_sched_barrier(0) between the fragment-load
// phase and the wmma phase so the machine scheduler cannot sink loads back
// into the wmma chain (round-2/3 asm showed it collapsing the batch into one
// register pair => load->wait(0)->wmma serialization).
// ---------------------------------------------------------------------------

typedef _Float16 half8 __attribute__((ext_vector_type(8)));
typedef _Float16 v16h  __attribute__((ext_vector_type(16)));
typedef float    v8f   __attribute__((ext_vector_type(8)));

#if defined(__has_builtin)
#if __has_builtin(__builtin_amdgcn_sched_barrier)
#define SCHED_FENCE() __builtin_amdgcn_sched_barrier(0)
#endif
#endif
#ifndef SCHED_FENCE
#define SCHED_FENCE()
#endif

#define Bn    2
#define Cn    32
#define Hn    2688
#define Wn    128
#define FIN   1345      // rfft bins of H=2688
#define KF    2720      // 2*FIN = 2690 padded to multiple of 32
#define MO    416       // 386 padded to multiple of 32 (13*32, 26*16)
#define OUTH  384
#define FOUT  193

__device__ __forceinline__ v8f wmma16(v16h a, v16h b, v8f c) {
  // D = A(16x32 f16) * B(32x16 f16) + C(16x16 f32)
  return __builtin_amdgcn_wmma_f32_16x16x32_f16(false, a, false, b, (short)0, c,
                                                false, false);
}

__device__ __forceinline__ v16h mkfrag(half8 lo, half8 hi) {
  v16h r;
#pragma unroll
  for (int i = 0; i < 8; ++i) { r[i] = lo[i]; r[i + 8] = hi[i]; }
  return r;
}

// A fragment: row-major matrix, two 8-half groups at +0 and +16 halves.
__device__ __forceinline__ v16h load_afrag(const _Float16* p) {
  return mkfrag(*(const half8*)p, *(const half8*)(p + 16));
}
// B fragment: two 8-half groups at +0 and +16 halves.
__device__ __forceinline__ v16h load_bfrag(const _Float16* p) {
  return mkfrag(*(const half8*)p, *(const half8*)(p + 16));
}

// ---------------- weight prep -------------------------------------------------

__global__ void k_prep_wspec(const float* __restrict__ wspec,
                             _Float16* __restrict__ wspec16) {
  int i = blockIdx.x * 256 + threadIdx.x;
  if (i >= MO * KF) return;
  int o = i / KF, f = i % KF;
  float v = (o < 386 && f < 2690) ? wspec[o * 2690 + f] : 0.f;
  wspec16[i] = (_Float16)v;
}

__global__ void k_prep_wsp(const float* __restrict__ wsp,
                           _Float16* __restrict__ wsp16) {
  int i = blockIdx.x * 256 + threadIdx.x;
  if (i >= 32 * 224) return;
  int co = i / 224, k = i % 224;
  int ci = k / 7, kh = k - ci * 7;
  wsp16[i] = (_Float16)wsp[(co * 32 + ci) * 7 + kh];
}

// M2[t][o]: ortho irfft(n=384) as a dense matrix; imag parts of bins 0 and 192
// are dropped (C2R semantics). Padded cols 386..415 are zero.
__global__ void k_build_m2(_Float16* __restrict__ m2) {
  int i = blockIdx.x * 256 + threadIdx.x;
  if (i >= OUTH * MO) return;
  int t = i / MO, o = i % MO;
  const float inv = 0.05103103630798288f;  // 1/sqrt(384)
  float v = 0.f;
  if (o < FOUT) {
    int k = o;
    if (k == 0)        v = inv;
    else if (k == 192) v = (t & 1) ? -inv : inv;
    else {
      int r = (k * t) % 384;
      v = 2.f * inv * __cosf((float)r * (6.283185307179586f / 384.f));
    }
  } else if (o < 386) {
    int k = o - FOUT;
    if (k >= 1 && k <= 191) {
      int r = (k * t) % 384;
      v = -2.f * inv * __sinf((float)r * (6.283185307179586f / 384.f));
    }
  }
  m2[i] = (_Float16)v;
}

// ---------------- M1 = W_spec @ D ---------------------------------------------
// M=416 (26 mtiles, split in 2 halves of 13), N=2688 (21 WGs of 128 cols),
// K=2720. D tiles generated on the fly into LDS (f16, [h][f] transposed).

__global__ __launch_bounds__(256, 1) void k_m1_gemm(
    const _Float16* __restrict__ wspec16, _Float16* __restrict__ m1) {
  __shared__ _Float16 ldsD[128 * 32];
  int h0 = blockIdx.x * 128;
  int mtbase = blockIdx.y * 13;
  int lane = threadIdx.x & 31;
  int wave = threadIdx.x >> 5;
  int n   = (wave << 4) + (lane & 15);
  int klo = (lane >> 4) << 3;   // 0 or 8  (A-frag K group offset / C row offset)
  int kbo = (lane >> 4) << 3;   // 0 or 8 (B-frag K offset)
  int hme = threadIdx.x & 127;
  int kk0 = (threadIdx.x >> 7) << 4;
  const float rs = 0.019287837f;  // 1/sqrt(2688)

  // Single A base; per-mtile displacement m*16*KF folds into the 24-bit ioffset.
  const _Float16* abase =
      wspec16 + (size_t)(mtbase * 16 + (lane & 15)) * KF + klo;

  v8f zero = {};
  v8f acc[13];
#pragma unroll
  for (int m = 0; m < 13; ++m) acc[m] = zero;

#pragma unroll 1
  for (int k0 = 0; k0 < KF; k0 += 32) {
    __syncthreads();
    int h = h0 + hme;
#pragma unroll
    for (int j = 0; j < 16; ++j) {
      int f = k0 + kk0 + j;
      float v = 0.f;
      if (f < 2690) {
        int fr = (f < FIN) ? f : (f - FIN);
        unsigned r = ((unsigned)(fr * h)) % 2688u;  // exact phase reduction
        float s, c;
        __sincosf((float)r * (6.283185307179586f / 2688.f), &s, &c);
        v = ((f < FIN) ? c : -s) * rs;
      }
      ldsD[hme * 32 + kk0 + j] = (_Float16)v;
    }
    __syncthreads();
    const _Float16* ak = abase + k0;
    // Phase 1: issue ALL fragment loads; keep them in flight together.
    v16h afr[13];
#pragma unroll
    for (int m = 0; m < 13; ++m)
      afr[m] = load_afrag(ak + (size_t)m * 16 * KF);
    v16h bf = load_bfrag(&ldsD[n * 32 + kbo]);
    SCHED_FENCE();  // forbid sinking loads into the wmma chain
    // Phase 2: 13 hazard-free back-to-back wmmas.
#pragma unroll
    for (int m = 0; m < 13; ++m) acc[m] = wmma16(afr[m], bf, acc[m]);
    SCHED_FENCE();
  }
  int hcol = h0 + n;
  _Float16* obase = m1 + (size_t)(mtbase * 16 + klo) * Hn + hcol;
#pragma unroll
  for (int m = 0; m < 13; ++m) {
#pragma unroll
    for (int e = 0; e < 8; ++e)
      obase[(size_t)(m * 16 + e) * Hn] = (_Float16)acc[m][e];
  }
}

// ---------------- spectral GEMM 1: Yt = relu(M1 @ X + b)^T --------------------
// Per (bc, mhalf): M=13x16 rows, N=128, K=2688. X chunk (32x128 fp32) staged
// transposed+f16 in LDS. Y stored transposed [bc][w][o] so GEMM-2 B-frags are
// contiguous b128 loads.

__global__ __launch_bounds__(256, 1) void k_spec_gemm1(
    const float* __restrict__ x, const _Float16* __restrict__ m1,
    const float* __restrict__ bspec, _Float16* __restrict__ yt) {
  __shared__ _Float16 ldsX[128 * 32];
  int bc = blockIdx.x;
  int mtbase = blockIdx.y * 13;
  int lane = threadIdx.x & 31;
  int wave = threadIdx.x >> 5;
  int n   = (wave << 4) + (lane & 15);
  int klo = (lane >> 4) << 3;
  int kbo = (lane >> 4) << 3;
  const float* xbc = x + (size_t)bc * Hn * Wn;
  int kk = threadIdx.x >> 3;          // 0..31
  int w0 = (threadIdx.x & 7) << 4;    // 0..112

  // Single A base; per-mtile displacement m*16*Hn folds into the ioffset.
  const _Float16* abase =
      m1 + (size_t)(mtbase * 16 + (lane & 15)) * Hn + klo;

  v8f zero = {};
  v8f acc[13];
#pragma unroll
  for (int m = 0; m < 13; ++m) acc[m] = zero;

#pragma unroll 1
  for (int k0 = 0; k0 < Hn; k0 += 32) {
    __syncthreads();
    const float* src = xbc + (size_t)(k0 + kk) * Wn + w0;
#pragma unroll
    for (int q4 = 0; q4 < 4; ++q4) {
      float4 f = *(const float4*)(src + q4 * 4);
      int wq = w0 + q4 * 4;
      ldsX[(wq + 0) * 32 + kk] = (_Float16)f.x;
      ldsX[(wq + 1) * 32 + kk] = (_Float16)f.y;
      ldsX[(wq + 2) * 32 + kk] = (_Float16)f.z;
      ldsX[(wq + 3) * 32 + kk] = (_Float16)f.w;
    }
    __syncthreads();
    const _Float16* ak = abase + k0;
    // Phase 1: issue ALL fragment loads; keep them in flight together.
    v16h afr[13];
#pragma unroll
    for (int m = 0; m < 13; ++m)
      afr[m] = load_afrag(ak + (size_t)m * 16 * Hn);
    v16h bf = load_bfrag(&ldsX[n * 32 + kbo]);
    SCHED_FENCE();  // forbid sinking loads into the wmma chain
    // Phase 2: 13 hazard-free back-to-back wmmas.
#pragma unroll
    for (int m = 0; m < 13; ++m) acc[m] = wmma16(afr[m], bf, acc[m]);
    SCHED_FENCE();
  }
  _Float16* ybase = yt + ((size_t)bc * 128 + n) * MO;
#pragma unroll
  for (int m = 0; m < 13; ++m) {
    int rowbase = (mtbase + m) * 16 + klo;
    half8 hv;
#pragma unroll
    for (int e = 0; e < 8; ++e) {
      int r = rowbase + e;
      float bv = (r < 386) ? bspec[r] : 0.f;
      float v = acc[m][e] + bv;
      hv[e] = (_Float16)(v > 0.f ? v : 0.f);
    }
    *(half8*)(ybase + rowbase) = hv;  // contiguous o, 16B store
  }
}

// ---------------- spatial branch: out = relu(Wsp @ Xslab + b) -----------------
// Per (t, b): [32x224] @ [224x128]; x slab staged transposed+f16 in LDS (56KB).
// 768 WGs -> occupancy hides latency; no sched fences needed.

__global__ __launch_bounds__(256, 1) void k_spatial(
    const float* __restrict__ x, const _Float16* __restrict__ wsp16,
    const float* __restrict__ bsp, float* __restrict__ out) {
  __shared__ _Float16 ldsS[128 * 224];
  int t = blockIdx.x;   // 0..383
  int b = blockIdx.y;   // 0..1
  int lane = threadIdx.x & 31;
  int wave = threadIdx.x >> 5;
  int n   = (wave << 4) + (lane & 15);
  int klo = (lane >> 4) << 3;
  int kbo = (lane >> 4) << 3;

  for (int v4 = threadIdx.x; v4 < 7168; v4 += 256) {
    int j = v4 << 2;
    int w = j & 127;
    int k = j >> 7;               // 0..223 == ci*7+kh
    int ci = k / 7, kh = k - ci * 7;
    const float* src =
        x + ((size_t)(b * 32 + ci) * Hn + 7 * t + kh) * Wn + w;
    float4 f = *(const float4*)src;
    ldsS[(w + 0) * 224 + k] = (_Float16)f.x;
    ldsS[(w + 1) * 224 + k] = (_Float16)f.y;
    ldsS[(w + 2) * 224 + k] = (_Float16)f.z;
    ldsS[(w + 3) * 224 + k] = (_Float16)f.w;
  }
  __syncthreads();

  v8f zero = {};
  v8f acc0 = zero, acc1 = zero;
  const _Float16* a0base = wsp16 + (size_t)(lane & 15) * 224 + klo;
#pragma unroll
  for (int kc = 0; kc < 224; kc += 32) {
    v16h bf  = load_bfrag(&ldsS[n * 224 + kc + kbo]);
    v16h af0 = load_afrag(a0base + kc);
    v16h af1 = load_afrag(a0base + kc + 16 * 224);
    acc0 = wmma16(af0, bf, acc0);
    acc1 = wmma16(af1, bf, acc1);
  }
#pragma unroll
  for (int e = 0; e < 8; ++e) {
    int co0 = klo + e;
    int co1 = 16 + klo + e;
    float v0 = acc0[e] + bsp[co0]; v0 = v0 > 0.f ? v0 : 0.f;
    float v1 = acc1[e] + bsp[co1]; v1 = v1 > 0.f ? v1 : 0.f;
    out[((size_t)(b * 32 + co0) * OUTH + t) * Wn + n] = v0;
    out[((size_t)(b * 32 + co1) * OUTH + t) * Wn + n] = v1;
  }
}

// ---------------- spectral GEMM 2: out += M2 @ Y ------------------------------
// Per (bc, t-tile): [16x416] @ [416x128]; both operands straight b128 loads,
// k-loop fully unrolled so all offsets are immediates. 1536 WGs -> occupancy
// hides latency.

__global__ __launch_bounds__(256, 1) void k_spec_gemm2(
    const _Float16* __restrict__ m2, const _Float16* __restrict__ yt,
    float* __restrict__ out) {
  int bc = blockIdx.x;  // 64
  int mt = blockIdx.y;  // 24
  int lane = threadIdx.x & 31;
  int wave = threadIdx.x >> 5;
  int n   = (wave << 4) + (lane & 15);
  int klo = (lane >> 4) << 3;
  int kbo = (lane >> 4) << 3;
  v8f acc = {};
  const _Float16* arow = m2 + (size_t)(mt * 16 + (lane & 15)) * MO + klo;
  const _Float16* brow = yt + ((size_t)bc * 128 + n) * MO + kbo;
#pragma unroll
  for (int k0 = 0; k0 < MO; k0 += 32) {
    v16h af = load_afrag(arow + k0);
    v16h bf = load_bfrag(brow + k0);
    acc = wmma16(af, bf, acc);
  }
  int t0 = mt * 16 + klo;
  float* op = out + ((size_t)bc * OUTH + t0) * Wn + n;
#pragma unroll
  for (int e = 0; e < 8; ++e) op[(size_t)e * Wn] += acc[e];
}

// ---------------- launch ------------------------------------------------------

extern "C" void kernel_launch(void* const* d_in, const int* in_sizes, int n_in,
                              void* d_out, int out_size, void* d_ws, size_t ws_size,
                              hipStream_t stream) {
  (void)in_sizes; (void)n_in; (void)out_size; (void)ws_size;
  const float* x     = (const float*)d_in[0];
  const float* wsp   = (const float*)d_in[1];
  const float* bsp   = (const float*)d_in[2];
  const float* wspec = (const float*)d_in[3];
  const float* bspec = (const float*)d_in[4];
  float* out = (float*)d_out;

  char* ws = (char*)d_ws;
  size_t off = 0;
  auto carve = [&](size_t bytes) -> void* {
    void* p = ws + off;
    off += (bytes + 255) & ~(size_t)255;
    return p;
  };
  _Float16* wspec16 = (_Float16*)carve((size_t)MO * KF * 2);      // 2.26 MB
  _Float16* m1      = (_Float16*)carve((size_t)MO * Hn * 2);      // 2.24 MB
  _Float16* m2      = (_Float16*)carve((size_t)OUTH * MO * 2);    // 0.32 MB
  _Float16* wsp16   = (_Float16*)carve((size_t)32 * 224 * 2);     // 14 KB
  _Float16* yt      = (_Float16*)carve((size_t)64 * 128 * MO * 2);// 6.8 MB

  k_prep_wspec<<<(MO * KF + 255) / 256, 256, 0, stream>>>(wspec, wspec16);
  k_prep_wsp<<<(32 * 224 + 255) / 256, 256, 0, stream>>>(wsp, wsp16);
  k_build_m2<<<(OUTH * MO + 255) / 256, 256, 0, stream>>>(m2);
  k_m1_gemm<<<dim3(Hn / 128, 2), 256, 0, stream>>>(wspec16, m1);
  k_spec_gemm1<<<dim3(Bn * Cn, 2), 256, 0, stream>>>(x, m1, bspec, yt);
  k_spatial<<<dim3(OUTH, Bn), 256, 0, stream>>>(x, wsp16, bsp, out);
  k_spec_gemm2<<<dim3(Bn * Cn, 24), 256, 0, stream>>>(m2, yt, out);
}
